// ThunderKittensMoE_75110388072961
// MI455X (gfx1250) — matrix operand order repaired and checked by reading, not compiled, not run on hardware
//
#include <hip/hip_runtime.h>

// MoE block for MI455X (gfx1250, wave32).
// Shapes fixed per reference: B=2 S=1024 -> N=2048 tokens, H=2048, I=1408, E=8, TOPK=2.
// All inputs fp32; expert math done in bf16 via v_wmma_f32_16x16x32_bf16 (f32 accum),
// matching the reference's bf16 expert computation. Routed experts computed densely
// (scaled by comb[n,e], zero for non-top-k), exactly like the reference einsums.

typedef __attribute__((ext_vector_type(16))) __bf16         v16bf;
typedef __attribute__((ext_vector_type(8)))  float          v8f;
typedef __attribute__((ext_vector_type(8)))  unsigned short v8u16;

#define MT 64          // token tile (M)
#define NT 64          // output-feature tile (N)
#define KT 32          // WMMA K per step (bf16)
#define LDS_STRIDE 40  // halfs per LDS row (80B) -> conflict-free b128 reads

union BFrag { v16bf v; v8u16 h[2]; };

__device__ __forceinline__ unsigned short f2bf(float f) {
  unsigned u = __float_as_uint(f);
  u += 0x7FFFu + ((u >> 16) & 1u);           // round-to-nearest-even
  return (unsigned short)(u >> 16);
}

// A fragment (16x32 bf16): lane l holds row M=l%16; lanes 0-15 carry K 0..7,16..23,
// lanes 16-31 carry K 8..15,24..31 (ISA 7.12.2 16-bit A layout).
__device__ __forceinline__ v16bf load_a_frag(const unsigned short* lds, int row) {
  const int lane = threadIdx.x & 31;
  const int kb = (lane < 16) ? 0 : 8;
  BFrag f;
  f.h[0] = *(const v8u16*)(lds + row * LDS_STRIDE + kb);
  f.h[1] = *(const v8u16*)(lds + row * LDS_STRIDE + kb + 16);
  return f.v;
}

// B fragment (32x16 bf16), operand staged as Bt rows (N x K): lane l holds col N=l%16;
// lanes 0-15 carry K 0..7,16..23, lanes 16-31 carry K 8..15,24..31 (same K order as A).
__device__ __forceinline__ v16bf load_b_frag(const unsigned short* lds, int row) {
  const int lane = threadIdx.x & 31;
  const int kb = (lane < 16) ? 0 : 8;
  BFrag f;
  f.h[0] = *(const v8u16*)(lds + row * LDS_STRIDE + kb);
  f.h[1] = *(const v8u16*)(lds + row * LDS_STRIDE + kb + 16);
  return f.v;
}

__device__ __forceinline__ v8f wmma_bf16(v16bf a, v16bf b, v8f c) {
  return __builtin_amdgcn_wmma_f32_16x16x32_bf16(false, a, false, b, (short)0, c,
                                                 false, false);
}

// ---------------------------------------------------------------- router ----
// one wave32 per token: logits over E=8, softmax, top-2 -> dense comb[N,8]
__global__ __launch_bounds__(256) void moe_router(const float* __restrict__ x,
                                                  const float* __restrict__ rw,
                                                  float* __restrict__ comb, int H) {
  const int wave = threadIdx.x >> 5;
  const int lane = threadIdx.x & 31;
  const int n = blockIdx.x * 8 + wave;
  const float* xr = x + (size_t)n * H;

  float acc[8] = {0.f, 0.f, 0.f, 0.f, 0.f, 0.f, 0.f, 0.f};
  for (int h = lane; h < H; h += 32) {
    const float xv = xr[h];
#pragma unroll
    for (int e = 0; e < 8; ++e) acc[e] += xv * rw[e * H + h];
  }
#pragma unroll
  for (int e = 0; e < 8; ++e) {
#pragma unroll
    for (int off = 16; off > 0; off >>= 1) acc[e] += __shfl_xor(acc[e], off, 32);
  }
  if (lane == 0) {
    float m = acc[0];
#pragma unroll
    for (int e = 1; e < 8; ++e) m = fmaxf(m, acc[e]);
    float w[8], s = 0.f;
#pragma unroll
    for (int e = 0; e < 8; ++e) { w[e] = __expf(acc[e] - m); s += w[e]; }
    const float inv = 1.0f / s;
#pragma unroll
    for (int e = 0; e < 8; ++e) w[e] *= inv;
    int i1 = 0;
#pragma unroll
    for (int e = 1; e < 8; ++e) if (w[e] > w[i1]) i1 = e;
    int i2 = (i1 == 0) ? 1 : 0;
#pragma unroll
    for (int e = 0; e < 8; ++e) if (e != i1 && w[e] > w[i2]) i2 = e;
#pragma unroll
    for (int e = 0; e < 8; ++e)
      comb[(size_t)n * 8 + e] = (e == i1) ? w[i1] : ((e == i2) ? w[i2] : 0.0f);
  }
}

// ------------------------------------------------------------ fp32 -> bf16 --
__global__ __launch_bounds__(256) void moe_cvt_bf16(const float* __restrict__ in,
                                                    unsigned short* __restrict__ out,
                                                    int n4) {
  const int i = blockIdx.x * blockDim.x + threadIdx.x;
  if (i < n4) {
    const float4 v = ((const float4*)in)[i];
    ushort4 o;
    o.x = f2bf(v.x); o.y = f2bf(v.y); o.z = f2bf(v.z); o.w = f2bf(v.w);
    ((ushort4*)out)[i] = o;
  }
}

// --------------------------------------------------- fused gate+up + sigmoid gate ---
// h[n,i] = sigmoid(x@Wg^T) * (x@Wu^T), bf16 out. Wg/Wu are [I,H] fp32, converted
// to bf16 while staging into LDS. blockIdx.z selects expert (stride 0 for shared).
__global__ __launch_bounds__(256) void moe_gateup(
    const unsigned short* __restrict__ xb,  // [N,H] bf16
    const float* __restrict__ wg_base, const float* __restrict__ wu_base,
    unsigned short* __restrict__ hout, int H, int I, long long wExpertStride,
    long long outExpertOff, long long outTokenStride) {
  __shared__ unsigned short lA[MT * LDS_STRIDE];
  __shared__ unsigned short lG[NT * LDS_STRIDE];
  __shared__ unsigned short lU[NT * LDS_STRIDE];

  const int ez = blockIdx.z;
  const float* wg = wg_base + (long long)ez * wExpertStride;
  const float* wu = wu_base + (long long)ez * wExpertStride;
  unsigned short* outb = hout + (long long)ez * outExpertOff;

  const int n0 = blockIdx.x * NT;  // I tile
  const int m0 = blockIdx.y * MT;  // token tile
  const int tid = threadIdx.x;
  const int wave = tid >> 5;
  const int lane = tid & 31;
  const int msub = wave & 3;   // M offset 16*msub
  const int nhalf = wave >> 2; // N offset 32*nhalf

  v8f aG0 = {}, aG1 = {}, aU0 = {}, aU1 = {};

  const int srow = tid >> 2;
  const int sc = (tid & 3) * 8;

  for (int k0 = 0; k0 < H; k0 += KT) {
    // ---- stage A (bf16 copy) and B tiles (fp32 -> bf16 convert) ----
    const unsigned short* ga = xb + (size_t)(m0 + srow) * H + k0 + sc;
    *(v8u16*)&lA[srow * LDS_STRIDE + sc] = *(const v8u16*)ga;

    const float* gg = wg + (size_t)(n0 + srow) * H + k0 + sc;
    const float* gu = wu + (size_t)(n0 + srow) * H + k0 + sc;
    {
      const float4 g0 = ((const float4*)gg)[0], g1 = ((const float4*)gg)[1];
      const float4 u0 = ((const float4*)gu)[0], u1 = ((const float4*)gu)[1];
      v8u16 bg, bu;
      bg[0] = f2bf(g0.x); bg[1] = f2bf(g0.y); bg[2] = f2bf(g0.z); bg[3] = f2bf(g0.w);
      bg[4] = f2bf(g1.x); bg[5] = f2bf(g1.y); bg[6] = f2bf(g1.z); bg[7] = f2bf(g1.w);
      bu[0] = f2bf(u0.x); bu[1] = f2bf(u0.y); bu[2] = f2bf(u0.z); bu[3] = f2bf(u0.w);
      bu[4] = f2bf(u1.x); bu[5] = f2bf(u1.y); bu[6] = f2bf(u1.z); bu[7] = f2bf(u1.w);
      *(v8u16*)&lG[srow * LDS_STRIDE + sc] = bg;
      *(v8u16*)&lU[srow * LDS_STRIDE + sc] = bu;
    }
    if (k0 + KT < H) {  // global_prefetch_b8 of next K tile
      __builtin_prefetch(ga + KT, 0, 0);
      __builtin_prefetch(gg + KT, 0, 0);
      __builtin_prefetch(gu + KT, 0, 0);
    }
    __syncthreads();

    const v16bf a = load_a_frag(lA, msub * 16 + (lane & 15));
    const int br = nhalf * 32 + (lane & 15);
    const v16bf bg0 = load_b_frag(lG, br);
    const v16bf bg1 = load_b_frag(lG, br + 16);
    const v16bf bu0 = load_b_frag(lU, br);
    const v16bf bu1 = load_b_frag(lU, br + 16);
    aG0 = wmma_bf16(a, bg0, aG0);
    aG1 = wmma_bf16(a, bg1, aG1);
    aU0 = wmma_bf16(a, bu0, aU0);
    aU1 = wmma_bf16(a, bu1, aU1);
    __syncthreads();
  }

  // ---- epilogue: h = sigmoid(g) * u, store bf16 ----
#pragma unroll
  for (int sub = 0; sub < 2; ++sub) {
    const v8f g = sub ? aG1 : aG0;
    const v8f u = sub ? aU1 : aU0;
#pragma unroll
    for (int i = 0; i < 8; ++i) {
      const int row = m0 + msub * 16 + i + ((lane >= 16) ? 8 : 0);
      const int col = n0 + nhalf * 32 + sub * 16 + (lane & 15);
      const float hv = u[i] * (1.0f / (1.0f + __expf(-g[i])));
      outb[(size_t)row * outTokenStride + col] = f2bf(hv);
    }
  }
}

// ------------------------------- down-proj + expert combine + residual ------
// out[n,h] = x[n,h] + Hs[n,:]@Wds[h,:] + sum_e comb[n,e] * He[n,e,:]@Wd[e][h,:]
__global__ __launch_bounds__(256) void moe_down(
    const unsigned short* __restrict__ hs,  // [N,I] bf16
    const unsigned short* __restrict__ he,  // [N,E,I] bf16
    const float* __restrict__ wds,          // [H,I]
    const float* __restrict__ wd,           // [E,H,I]
    const float* __restrict__ comb,         // [N,E]
    const float* __restrict__ x,            // [N,H]
    float* __restrict__ out, int H, int I) {
  constexpr int NE = 8;
  __shared__ unsigned short lA[MT * LDS_STRIDE];
  __shared__ unsigned short lB[NT * LDS_STRIDE];
  __shared__ float sComb[MT];

  const int n0 = blockIdx.x * NT;  // H tile
  const int m0 = blockIdx.y * MT;  // token tile
  const int tid = threadIdx.x;
  const int wave = tid >> 5;
  const int lane = tid & 31;
  const int msub = wave & 3;
  const int nhalf = wave >> 2;
  const int srow = tid >> 2;
  const int sc = (tid & 3) * 8;

  v8f f0 = {}, f1 = {};

  for (int p = 0; p < 1 + NE; ++p) {
    const int e = p - 1;
    __syncthreads();  // WAR on sComb vs previous phase's epilogue reads
    if (p > 0 && tid < MT) sComb[tid] = comb[(size_t)(m0 + tid) * NE + e];
    __syncthreads();

    bool skip = false;
    if (p > 0) {
      bool nz = false;
      for (int j = 0; j < MT; ++j) nz |= (sComb[j] != 0.0f);
      skip = !nz;  // uniform across block
    }
    if (skip) continue;

    const float* wbase = (p == 0) ? wds : (wd + (size_t)e * H * I);
    v8f p0 = {}, p1 = {};

    for (int k0 = 0; k0 < I; k0 += KT) {
      const unsigned short* ga =
          (p == 0) ? (hs + (size_t)(m0 + srow) * I + k0 + sc)
                   : (he + ((size_t)(m0 + srow) * NE + e) * I + k0 + sc);
      *(v8u16*)&lA[srow * LDS_STRIDE + sc] = *(const v8u16*)ga;

      const float* gw = wbase + (size_t)(n0 + srow) * I + k0 + sc;
      {
        const float4 w0 = ((const float4*)gw)[0], w1 = ((const float4*)gw)[1];
        v8u16 bw;
        bw[0] = f2bf(w0.x); bw[1] = f2bf(w0.y); bw[2] = f2bf(w0.z); bw[3] = f2bf(w0.w);
        bw[4] = f2bf(w1.x); bw[5] = f2bf(w1.y); bw[6] = f2bf(w1.z); bw[7] = f2bf(w1.w);
        *(v8u16*)&lB[srow * LDS_STRIDE + sc] = bw;
      }
      if (k0 + KT < I) {
        __builtin_prefetch(ga + KT, 0, 0);
        __builtin_prefetch(gw + KT, 0, 0);
      }
      __syncthreads();

      const v16bf a = load_a_frag(lA, msub * 16 + (lane & 15));
      const int br = nhalf * 32 + (lane & 15);
      const v16bf b0 = load_b_frag(lB, br);
      const v16bf b1 = load_b_frag(lB, br + 16);
      p0 = wmma_bf16(a, b0, p0);
      p1 = wmma_bf16(a, b1, p1);
      __syncthreads();
    }

    // scale partials by comb[n,e] (1.0 for shared) and accumulate
#pragma unroll
    for (int i = 0; i < 8; ++i) {
      const int rl = msub * 16 + i + ((lane >= 16) ? 8 : 0);
      const float s = (p == 0) ? 1.0f : sComb[rl];
      f0[i] += s * p0[i];
      f1[i] += s * p1[i];
    }
  }

  // residual + store fp32
#pragma unroll
  for (int i = 0; i < 8; ++i) {
    const int row = m0 + msub * 16 + i + ((lane >= 16) ? 8 : 0);
    const int col = n0 + nhalf * 32 + (lane & 15);
    out[(size_t)row * H + col] = x[(size_t)row * H + col] + f0[i];
    out[(size_t)row * H + col + 16] = x[(size_t)row * H + col + 16] + f1[i];
  }
}

// ----------------------------------------------------------------------------
extern "C" void kernel_launch(void* const* d_in, const int* in_sizes, int n_in,
                              void* d_out, int out_size, void* d_ws, size_t ws_size,
                              hipStream_t stream) {
  (void)in_sizes; (void)n_in; (void)out_size; (void)ws_size;
  constexpr int N = 2048, H = 2048, I = 1408, E = 8;

  const float* x           = (const float*)d_in[0];
  const float* router_w    = (const float*)d_in[1];
  const float* shared_gate = (const float*)d_in[2];
  const float* shared_up   = (const float*)d_in[3];
  const float* shared_down = (const float*)d_in[4];
  const float* w_gate      = (const float*)d_in[5];
  const float* w_up        = (const float*)d_in[6];
  const float* w_down      = (const float*)d_in[7];
  float* out = (float*)d_out;

  char* ws = (char*)d_ws;
  float* comb = (float*)ws;                                       // N*E*4   = 64 KB
  unsigned short* xb = (unsigned short*)(ws + 65536);             // N*H*2   = 8 MB
  unsigned short* hs = (unsigned short*)(ws + 65536 + (size_t)N * H * 2);   // N*I*2
  unsigned short* he =
      (unsigned short*)(ws + 65536 + (size_t)N * H * 2 + (size_t)N * I * 2); // N*E*I*2

  moe_router<<<N / 8, 256, 0, stream>>>(x, router_w, comb, H);
  moe_cvt_bf16<<<(N * H / 4 + 255) / 256, 256, 0, stream>>>(x, xb, N * H / 4);

  dim3 gShared(I / NT, N / MT, 1);
  moe_gateup<<<gShared, 256, 0, stream>>>(xb, shared_gate, shared_up, hs, H, I,
                                          0LL, 0LL, (long long)I);
  dim3 gRouted(I / NT, N / MT, E);
  moe_gateup<<<gRouted, 256, 0, stream>>>(xb, w_gate, w_up, he, H, I,
                                          (long long)I * H, (long long)I,
                                          (long long)E * I);
  dim3 gDown(H / NT, N / MT, 1);
  moe_down<<<gDown, 256, 0, stream>>>(hs, he, shared_down, w_down, comb, x, out, H, I);
}
